// EncoderLayer_7292854468648
// MI455X (gfx1250) — matrix operand order repaired and checked by reading, not compiled, not run on hardware
//
#include <hip/hip_runtime.h>
#include <hip/hip_bf16.h>
#include <math.h>

// ---------------------------------------------------------------------------
// Types / helpers
// ---------------------------------------------------------------------------
typedef __bf16 bf16_t;
typedef __attribute__((ext_vector_type(8)))  bf16_t v8bf;
typedef __attribute__((ext_vector_type(16))) bf16_t v16bf;
typedef __attribute__((ext_vector_type(8)))  float  v8f;
typedef __attribute__((ext_vector_type(4)))  unsigned v4u;
typedef __attribute__((ext_vector_type(8)))  int    v8i;
typedef __attribute__((ext_vector_type(4)))  int    v4i;

#define DEV __device__ __forceinline__

// Staging-path selection (probe-confirmed: TDM path active on this toolchain).
#if __has_builtin(__builtin_amdgcn_tensor_load_to_lds)
#define PATH_TDM 1
#define PATH_ASYNC 0
#elif __has_builtin(__builtin_amdgcn_global_load_async_to_lds_b128)
#define PATH_TDM 0
#define PATH_ASYNC 1
#else
#define PATH_TDM 0
#define PATH_ASYNC 0
#endif

DEV bf16_t f2bf(float f) {
  union { float f; unsigned u; } x; x.f = f;
  unsigned r = x.u + 0x7FFFu + ((x.u >> 16) & 1u);     // round-to-nearest-even
  unsigned short h = (unsigned short)(r >> 16);
  return __builtin_bit_cast(bf16_t, h);
}

DEV v8f zero8() {
  v8f z;
#pragma unroll
  for (int e = 0; e < 8; ++e) z[e] = 0.0f;
  return z;
}

// A-fragment (16x32 bf16): lane holds row = lane&15; elements 0..7 = K kb..kb+7,
// elements 8..15 = K kb+16..kb+23, kb = 8*(lane>>4).   (ISA 7.12.2, 16-bit A)
DEV v16bf load_frag_a(const bf16_t* base, int ld) {
  int lane = threadIdx.x & 31;
  int r  = lane & 15;
  int kb = (lane >> 4) * 8;
  const bf16_t* p = base + (size_t)r * ld + kb;
  v8bf lo = *(const v8bf*)(p);
  v8bf hi = *(const v8bf*)(p + 16);
  v16bf a;
#pragma unroll
  for (int i = 0; i < 8; ++i) { a[i] = lo[i]; a[i + 8] = hi[i]; }
  return a;
}

// B-fragment (32x16 bf16, KxN): lane holds column = lane&15; elements 0..7 = K kb..kb+7,
// elements 8..15 = K kb+16..kb+23, kb = 8*(lane>>4).   (matches A fragment)
DEV v16bf load_frag_b(const bf16_t* base, int ld) {
  int lane = threadIdx.x & 31;
  int c  = lane & 15;
  int kb = (lane >> 4) * 8;
  const bf16_t* p = base + (size_t)c * ld + kb;
  v8bf lo = *(const v8bf*)(p);
  v8bf hi = *(const v8bf*)(p + 16);
  v16bf b;
#pragma unroll
  for (int i = 0; i < 8; ++i) { b[i] = lo[i]; b[i + 8] = hi[i]; }
  return b;
}

DEV v8f wmma_bf16(v16bf a, v16bf b, v8f c) {
  return __builtin_amdgcn_wmma_f32_16x16x32_bf16(false, a, false, b,
                                                 (short)0, c, false, false);
}

// ---------------------------------------------------------------------------
// 16-lane crossbar reductions: v_permlane16_b32 (pure VALU, no LDS/dscnt).
// ---------------------------------------------------------------------------
#if __has_builtin(__builtin_amdgcn_permlane16)
DEV float permx(float v, unsigned sl, unsigned sh) {
  int i = __builtin_bit_cast(int, v);
  int r = __builtin_amdgcn_permlane16(i, i, sl, sh, false, false);
  return __builtin_bit_cast(float, r);
}
DEV float rmax16(float v) {
  v = fmaxf(v, permx(v, 0x67452301u, 0xEFCDAB89u));   // xor 1
  v = fmaxf(v, permx(v, 0x54761032u, 0xDCFE98BAu));   // xor 2
  v = fmaxf(v, permx(v, 0x32107654u, 0xBA98FEDCu));   // xor 4
  v = fmaxf(v, permx(v, 0xFEDCBA98u, 0x76543210u));   // xor 8
  return v;
}
DEV float radd16(float v) {
  v += permx(v, 0x67452301u, 0xEFCDAB89u);
  v += permx(v, 0x54761032u, 0xDCFE98BAu);
  v += permx(v, 0x32107654u, 0xBA98FEDCu);
  v += permx(v, 0xFEDCBA98u, 0x76543210u);
  return v;
}
#else
DEV float rmax16(float v) {
  v = fmaxf(v, __shfl_xor(v, 1, 32));
  v = fmaxf(v, __shfl_xor(v, 2, 32));
  v = fmaxf(v, __shfl_xor(v, 4, 32));
  v = fmaxf(v, __shfl_xor(v, 8, 32));
  return v;
}
DEV float radd16(float v) {
  v += __shfl_xor(v, 1, 32);
  v += __shfl_xor(v, 2, 32);
  v += __shfl_xor(v, 4, 32);
  v += __shfl_xor(v, 8, 32);
  return v;
}
#endif

// ---------------------------------------------------------------------------
// TDM path: async DMA of a 2-D tile (rows x 32 bf16) global -> LDS, with HW
// padding of each 64B row by 16B so LDS row stride = 40 bf16.
// D# layout per CDNA5 ISA.
// ---------------------------------------------------------------------------
#if PATH_TDM
DEV void tdm_load_tile_2d(unsigned lds_off, const void* gptr,
                          unsigned tensor_d0, unsigned tensor_d1,
                          unsigned tile_d0, unsigned tile_d1,
                          unsigned stride_elems) {
  unsigned long long ga = (unsigned long long)(uintptr_t)gptr;
  v4u g0;
  g0[0] = 1u;                                             // count=1, user D#
  g0[1] = lds_off;                                        // lds_addr (bytes)
  g0[2] = (unsigned)(ga & 0xFFFFFFFFu);                   // global_addr lo
  g0[3] = (unsigned)((ga >> 32) & 0x01FFFFFFu) | 0x80000000u; // hi | type=2
  v8i g1;
  // data_size=2B (code 1), pad_enable, pad_interval=16 DW (3), pad_amount=4 DW (3)
  g1[0] = (int)((1u << 16) | (1u << 20) | (3u << 22) | (3u << 25));
  g1[1] = (int)((tensor_d0 & 0xFFFFu) << 16);             // tensor_dim0 lo16
  g1[2] = (int)(((tensor_d0 >> 16) & 0xFFFFu) | ((tensor_d1 & 0xFFFFu) << 16));
  g1[3] = (int)(((tensor_d1 >> 16) & 0xFFFFu) | (tile_d0 << 16));
  g1[4] = (int)(tile_d1 & 0xFFFFu);                       // tile_dim1, tile_dim2=0
  g1[5] = (int)stride_elems;                              // dim0_stride lo32
  g1[6] = 0;
  g1[7] = 0;
  v4i gz; gz[0] = 0; gz[1] = 0; gz[2] = 0; gz[3] = 0;
#if defined(__clang_major__) && (__clang_major__ >= 23)
  v8i gz8;
#pragma unroll
  for (int i = 0; i < 8; ++i) gz8[i] = 0;
  __builtin_amdgcn_tensor_load_to_lds(g0, g1, gz, gz, gz8, 0);
#else
  __builtin_amdgcn_tensor_load_to_lds(g0, g1, gz, gz, 0);
#endif
}

DEV unsigned lds_offset_of(const void* p) {
  return (unsigned)(uintptr_t)p;   // LDS aperture: addr[31:0] = LDS byte offset
}
#endif

// ---------------------------------------------------------------------------
// ASYNC fallback path: per-lane 16B async global -> LDS (ASYNCcnt tracked).
// ---------------------------------------------------------------------------
#if PATH_ASYNC
typedef __attribute__((address_space(1))) void* as1p;
typedef __attribute__((address_space(3))) void* as3p;
DEV void async_copy16(bf16_t* lds, const bf16_t* g) {
  as1p gp = (as1p)(unsigned long long)(uintptr_t)g;
  as3p lp = (as3p)(unsigned)(uintptr_t)lds;
  __builtin_amdgcn_global_load_async_to_lds_b128(gp, lp, 0, 0);
}
DEV void wait_async0() {
#if __has_builtin(__builtin_amdgcn_s_wait_asynccnt)
  __builtin_amdgcn_s_wait_asynccnt(0);
#else
  asm volatile("s_wait_asynccnt 0x0" ::: "memory");
#endif
}
#endif

// ---------------------------------------------------------------------------
// Elementwise converts
// ---------------------------------------------------------------------------
__global__ void f32_to_bf16_kernel(const float* __restrict__ in,
                                   bf16_t* __restrict__ out, int n) {
  int i = blockIdx.x * 256 + threadIdx.x;
  if (i < n) out[i] = f2bf(in[i]);
}

// W [K,N] f32 row-major  ->  Wt [N,K] bf16 row-major
__global__ void transpose_bf16_kernel(const float* __restrict__ W,
                                      bf16_t* __restrict__ Wt, int K, int N) {
  int i = blockIdx.x * 256 + threadIdx.x;
  if (i >= N * K) return;
  int n = i / K, k = i % K;
  Wt[i] = f2bf(W[(size_t)k * N + n]);
}

// ---------------------------------------------------------------------------
// bf16 GEMM:  C[M,N] = A[M,K] @ Wt[N,K]^T + bias, with epilogue variants.
// 256 threads = 8 waves; block tile 128x128, BK=32; wave tile 32x64.
// Tiles staged global->LDS by the Tensor Data Mover, double-buffered on
// TENSORcnt; wave 0 drives the DMA, all 8 waves consume via WMMA.
// EPI: 0 = f32 out[M,N]; 2 = exact GELU -> bf16 out[M,N];
//      3 = bf16 out laid out [B,H,S,HD] (Q/K); 4 = bf16 out [B,H,HD,S] (V^T)
// ---------------------------------------------------------------------------
template <int EPI>
__global__ __launch_bounds__(256)
void gemm_bf16_kernel(const bf16_t* __restrict__ A, const bf16_t* __restrict__ Wt,
                      const float* __restrict__ bias, void* __restrict__ outp,
                      int M, int N, int K, int S, int H, int HD) {
  const int LD = 40;                       // padded LDS stride (bf16 elems)
  __shared__ bf16_t As[2][128 * LD];
  __shared__ bf16_t Bs[2][128 * LD];

  int tid  = threadIdx.x;
  int wave = tid >> 5;
  int wm   = wave & 3;                     // 4 waves along M
  int wn   = wave >> 2;                    // 2 waves along N
  int m0 = blockIdx.y * 128, n0 = blockIdx.x * 128;

  v8f acc[2][4];
#pragma unroll
  for (int i = 0; i < 2; ++i)
#pragma unroll
    for (int j = 0; j < 4; ++j) acc[i][j] = zero8();

  const int nsteps = K >> 5;

#if PATH_TDM
  if (wave == 0) {                         // prime buffer 0
    tdm_load_tile_2d(lds_offset_of(&As[0][0]), &A[(size_t)m0 * K],
                     (unsigned)K, 128u, 32u, 128u, (unsigned)K);
    tdm_load_tile_2d(lds_offset_of(&Bs[0][0]), &Wt[(size_t)n0 * K],
                     (unsigned)K, 128u, 32u, 128u, (unsigned)K);
  }
#endif

  for (int s = 0; s < nsteps; ++s) {
    int cur = s & 1;
#if PATH_TDM
    if (wave == 0) {
      if (s + 1 < nsteps) {                // prefetch next tile into other buffer
        int k0n = (s + 1) * 32;
        tdm_load_tile_2d(lds_offset_of(&As[cur ^ 1][0]),
                         &A[(size_t)m0 * K + k0n],
                         (unsigned)K, 128u, 32u, 128u, (unsigned)K);
        tdm_load_tile_2d(lds_offset_of(&Bs[cur ^ 1][0]),
                         &Wt[(size_t)n0 * K + k0n],
                         (unsigned)K, 128u, 32u, 128u, (unsigned)K);
        __builtin_amdgcn_s_wait_tensorcnt(2);   // current tile's pair done
      } else {
        __builtin_amdgcn_s_wait_tensorcnt(0);
      }
    }
    __syncthreads();
#elif PATH_ASYNC
    int k0 = s * 32;
#pragma unroll
    for (int it = 0; it < 2; ++it) {       // 512 16B chunks over 256 lanes
      int chunk = tid + it * 256;
      int row = chunk >> 2;
      int col = (chunk & 3) * 8;
      async_copy16(&As[cur][row * LD + col], &A[(size_t)(m0 + row) * K + k0 + col]);
      async_copy16(&Bs[cur][row * LD + col], &Wt[(size_t)(n0 + row) * K + k0 + col]);
    }
    wait_async0();
    __syncthreads();
#else
    int k0 = s * 32;
#pragma unroll
    for (int it = 0; it < 2; ++it) {       // 512 v8bf chunks over 256 threads
      int chunk = tid + it * 256;
      int row = chunk >> 2;
      int col = (chunk & 3) * 8;
      *(v8bf*)&As[cur][row * LD + col] =
          *(const v8bf*)&A[(size_t)(m0 + row) * K + k0 + col];
      *(v8bf*)&Bs[cur][row * LD + col] =
          *(const v8bf*)&Wt[(size_t)(n0 + row) * K + k0 + col];
    }
    __syncthreads();
#endif

    v16bf af[2], bfr[4];
#pragma unroll
    for (int i = 0; i < 2; ++i)
      af[i] = load_frag_a(&As[cur][(wm * 32 + i * 16) * LD], LD);
#pragma unroll
    for (int j = 0; j < 4; ++j)
      bfr[j] = load_frag_b(&Bs[cur][(wn * 64 + j * 16) * LD], LD);
#pragma unroll
    for (int i = 0; i < 2; ++i)
#pragma unroll
      for (int j = 0; j < 4; ++j) acc[i][j] = wmma_bf16(af[i], bfr[j], acc[i][j]);
    __syncthreads();                       // buffer reuse fence
  }

  int lane = tid & 31;
  int hi = lane >> 4, lc = lane & 15;
#pragma unroll
  for (int i = 0; i < 2; ++i) {
#pragma unroll
    for (int j = 0; j < 4; ++j) {
#pragma unroll
      for (int e = 0; e < 8; ++e) {
        int m = m0 + wm * 32 + i * 16 + e + 8 * hi;   // C layout: VGPR e -> row
        int n = n0 + wn * 64 + j * 16 + lc;           // lane&15 -> column
        float v = acc[i][j][e] + bias[n];
        if (EPI == 0) {
          ((float*)outp)[(size_t)m * N + n] = v;
        } else if (EPI == 2) {
          float g = 0.5f * v * (1.0f + erff(v * 0.70710678118654752f));
          ((bf16_t*)outp)[(size_t)m * N + n] = f2bf(g);
        } else if (EPI == 3) {
          int b = m / S, s2 = m % S, h = n / HD, hd = n % HD;
          ((bf16_t*)outp)[(((size_t)b * H + h) * S + s2) * HD + hd] = f2bf(v);
        } else if (EPI == 4) {
          int b = m / S, s2 = m % S, h = n / HD, hd = n % HD;
          ((bf16_t*)outp)[(((size_t)b * H + h) * HD + hd) * S + s2] = f2bf(v);
        }
      }
    }
  }
}

// ---------------------------------------------------------------------------
// Flash attention: one wave (32 threads) per 16-query tile per (b,h).
// Q,K in [B,H,S,HD] bf16; V in [B,H,HD,S] bf16 (transposed); out attn [B,S,D] bf16.
// ---------------------------------------------------------------------------
__global__ __launch_bounds__(32)
void attn_kernel(const bf16_t* __restrict__ Q, const bf16_t* __restrict__ Kt,
                 const bf16_t* __restrict__ Vt, const float* __restrict__ mask,
                 bf16_t* __restrict__ out, int S, int H, int HD, int D) {
  const int LD = 40;
  __shared__ bf16_t Pt[16 * LD];           // prob tile 16x32 (transposed staging)

  int qtiles = S / 16;
  int bh = blockIdx.x / qtiles;
  int qt = blockIdx.x % qtiles;
  int b = bh / H, h = bh % H;
  size_t base = (size_t)bh * S * HD;
  int lane = threadIdx.x & 31;
  int hi = lane >> 4, lc = lane & 15;
  const float scale = rsqrtf((float)HD);

  v16bf aq0 = load_frag_a(Q + base + (size_t)(qt * 16) * HD,      HD); // hd 0..31
  v16bf aq1 = load_frag_a(Q + base + (size_t)(qt * 16) * HD + 32, HD); // hd 32..63

  float mrow[8], lrow[8];
  v8f o[4];
#pragma unroll
  for (int e = 0; e < 8; ++e) { mrow[e] = -1e30f; lrow[e] = 0.0f; }
#pragma unroll
  for (int n = 0; n < 4; ++n) o[n] = zero8();

  for (int key0 = 0; key0 < S; key0 += 32) {
    const bf16_t* K0 = Kt + base + (size_t)key0 * HD;
    const bf16_t* K1 = K0 + 16 * HD;
    v8f s0 = zero8(), s1 = zero8();
    s0 = wmma_bf16(aq0, load_frag_b(K0,      HD), s0);
    s0 = wmma_bf16(aq1, load_frag_b(K0 + 32, HD), s0);
    s1 = wmma_bf16(aq0, load_frag_b(K1,      HD), s1);
    s1 = wmma_bf16(aq1, load_frag_b(K1 + 32, HD), s1);

    float mv0 = mask[(size_t)b * S + key0 + lc];
    float mv1 = mask[(size_t)b * S + key0 + 16 + lc];

    float p0[8], p1[8];
#pragma unroll
    for (int e = 0; e < 8; ++e) {
      float t0 = s0[e] * scale + mv0;
      float t1 = s1[e] * scale + mv1;
      float rm = rmax16(fmaxf(t0, t1));                 // row max (16-lane xbar)
      float mn = fmaxf(mrow[e], rm);
      float alpha = __expf(mrow[e] - mn);
      mrow[e] = mn;
      float e0 = __expf(t0 - mn), e1 = __expf(t1 - mn);
      float rs = radd16(e0 + e1);                       // row sum
      lrow[e] = lrow[e] * alpha + rs;
      p0[e] = e0; p1[e] = e1;
#pragma unroll
      for (int n = 0; n < 4; ++n) o[n][e] *= alpha;
    }

    __syncthreads();                        // single wave: LDS in-order anyway
#pragma unroll
    for (int e = 0; e < 8; ++e) {           // C layout -> row-major 16x32 tile
      Pt[(e + 8 * hi) * LD + lc]      = f2bf(p0[e]);
      Pt[(e + 8 * hi) * LD + lc + 16] = f2bf(p1[e]);
    }
    __syncthreads();

    v16bf ap = load_frag_a(Pt, LD);         // re-read as A fragment (16x32 keys)
#pragma unroll
    for (int n = 0; n < 4; ++n) {
      const bf16_t* Vp = Vt + (size_t)bh * HD * S + (size_t)(n * 16) * S + key0;
      o[n] = wmma_bf16(ap, load_frag_b(Vp, S), o[n]);
    }
  }

#pragma unroll
  for (int n = 0; n < 4; ++n)
#pragma unroll
    for (int e = 0; e < 8; ++e) {
      int srow = qt * 16 + e + 8 * hi;
      int col  = h * HD + n * 16 + lc;
      out[((size_t)b * S + srow) * D + col] = f2bf(o[n][e] / lrow[e]);
    }
}

// ---------------------------------------------------------------------------
// Fused residual-add + LayerNorm (f32), emits f32 and optional bf16 copies.
// One 256-thread block per row of D=1024.
// ---------------------------------------------------------------------------
__global__ __launch_bounds__(256)
void ln_kernel(const float* __restrict__ resid, const float* __restrict__ y,
               const float* __restrict__ g, const float* __restrict__ be,
               float* __restrict__ outf, bf16_t* __restrict__ outb, int D) {
  int row = blockIdx.x;
  const float* r = resid + (size_t)row * D;
  const float* q = y + (size_t)row * D;
  int tid = threadIdx.x;

  float s = 0.0f, s2 = 0.0f;
  for (int i = tid; i < D; i += 256) {
    float v = r[i] + q[i];
    s += v; s2 += v * v;
  }
  for (int m = 1; m < 32; m <<= 1) {
    s  += __shfl_xor(s,  m, 32);
    s2 += __shfl_xor(s2, m, 32);
  }
  __shared__ float sh[16];
  int wave = tid >> 5;
  if ((tid & 31) == 0) { sh[wave] = s; sh[8 + wave] = s2; }
  __syncthreads();
  float ts = 0.0f, ts2 = 0.0f;
#pragma unroll
  for (int wv = 0; wv < 8; ++wv) { ts += sh[wv]; ts2 += sh[8 + wv]; }

  float mu   = ts / D;
  float var  = ts2 / D - mu * mu;
  float rstd = rsqrtf(var + 1e-5f);
  for (int i = tid; i < D; i += 256) {
    float v = r[i] + q[i];
    float o = (v - mu) * rstd * g[i] + be[i];
    outf[(size_t)row * D + i] = o;
    if (outb) outb[(size_t)row * D + i] = f2bf(o);
  }
}

// ---------------------------------------------------------------------------
// Launch
// ---------------------------------------------------------------------------
extern "C" void kernel_launch(void* const* d_in, const int* in_sizes, int n_in,
                              void* d_out, int out_size, void* d_ws, size_t ws_size,
                              hipStream_t stream) {
  (void)in_sizes; (void)n_in; (void)out_size; (void)ws_size;
  const int Bb = 2, S = 2048, D = 1024, H = 16, HD = 64, F = 4096;
  const int M = Bb * S;

  const float* x    = (const float*)d_in[0];
  const float* mask = (const float*)d_in[1];
  const float* Wq   = (const float*)d_in[2];  const float* bq = (const float*)d_in[3];
  const float* Wk   = (const float*)d_in[4];  const float* bk = (const float*)d_in[5];
  const float* Wv   = (const float*)d_in[6];  const float* bv = (const float*)d_in[7];
  const float* Wo   = (const float*)d_in[8];  const float* bo = (const float*)d_in[9];
  const float* W1   = (const float*)d_in[10]; const float* b1 = (const float*)d_in[11];
  const float* W2   = (const float*)d_in[12]; const float* b2 = (const float*)d_in[13];
  const float* ga1  = (const float*)d_in[14]; const float* be1 = (const float*)d_in[15];
  const float* ga2  = (const float*)d_in[16]; const float* be2 = (const float*)d_in[17];

  char* wp = (char*)d_ws;
  auto alloc = [&](size_t bytes) -> void* {
    void* p = (void*)wp;
    wp += (bytes + 255) & ~(size_t)255;
    return p;
  };
  bf16_t* xb    = (bf16_t*)alloc((size_t)M * D * 2);
  bf16_t* WqT   = (bf16_t*)alloc((size_t)D * D * 2);
  bf16_t* WkT   = (bf16_t*)alloc((size_t)D * D * 2);
  bf16_t* WvT   = (bf16_t*)alloc((size_t)D * D * 2);
  bf16_t* WoT   = (bf16_t*)alloc((size_t)D * D * 2);
  bf16_t* W1T   = (bf16_t*)alloc((size_t)D * F * 2);
  bf16_t* W2T   = (bf16_t*)alloc((size_t)F * D * 2);
  bf16_t* Qb    = (bf16_t*)alloc((size_t)M * D * 2);
  bf16_t* Kb    = (bf16_t*)alloc((size_t)M * D * 2);
  bf16_t* Vtb   = (bf16_t*)alloc((size_t)M * D * 2);
  bf16_t* attnb = (bf16_t*)alloc((size_t)M * D * 2);
  float*  y1    = (float*) alloc((size_t)M * D * 4);
  float*  x1f   = (float*) alloc((size_t)M * D * 4);
  bf16_t* x1b   = (bf16_t*)alloc((size_t)M * D * 2);
  bf16_t* hb    = (bf16_t*)alloc((size_t)M * F * 2);
  float*  y2    = (float*) alloc((size_t)M * D * 4);

  // bf16 conversions / weight transposes
  f32_to_bf16_kernel<<<(M * D + 255) / 256, 256, 0, stream>>>(x, xb, M * D);
  transpose_bf16_kernel<<<(D * D + 255) / 256, 256, 0, stream>>>(Wq, WqT, D, D);
  transpose_bf16_kernel<<<(D * D + 255) / 256, 256, 0, stream>>>(Wk, WkT, D, D);
  transpose_bf16_kernel<<<(D * D + 255) / 256, 256, 0, stream>>>(Wv, WvT, D, D);
  transpose_bf16_kernel<<<(D * D + 255) / 256, 256, 0, stream>>>(Wo, WoT, D, D);
  transpose_bf16_kernel<<<(D * F + 255) / 256, 256, 0, stream>>>(W1, W1T, D, F);
  transpose_bf16_kernel<<<(F * D + 255) / 256, 256, 0, stream>>>(W2, W2T, F, D);

  dim3 gD(D / 128, M / 128);   // N=1024 GEMMs
  dim3 gF(F / 128, M / 128);   // N=4096 GEMM

  // QKV projections (epilogues scatter into attention-friendly layouts)
  gemm_bf16_kernel<3><<<gD, 256, 0, stream>>>(xb, WqT, bq, Qb,  M, D, D, S, H, HD);
  gemm_bf16_kernel<3><<<gD, 256, 0, stream>>>(xb, WkT, bk, Kb,  M, D, D, S, H, HD);
  gemm_bf16_kernel<4><<<gD, 256, 0, stream>>>(xb, WvT, bv, Vtb, M, D, D, S, H, HD);

  // Flash attention
  attn_kernel<<<Bb * H * (S / 16), 32, 0, stream>>>(Qb, Kb, Vtb, mask, attnb,
                                                    S, H, HD, D);

  // Output projection, residual + LN1
  gemm_bf16_kernel<0><<<gD, 256, 0, stream>>>(attnb, WoT, bo, y1, M, D, D, S, H, HD);
  ln_kernel<<<M, 256, 0, stream>>>(x, y1, ga1, be1, x1f, x1b, D);

  // FFN (GELU fused into GEMM1 epilogue), residual + LN2 -> d_out
  gemm_bf16_kernel<2><<<gF, 256, 0, stream>>>(x1b, W1T, b1, hb, M, F, D, S, H, HD);
  gemm_bf16_kernel<0><<<gD, 256, 0, stream>>>(hb, W2T, b2, y2, M, D, F, S, H, HD);
  ln_kernel<<<M, 256, 0, stream>>>(x1f, y2, ga2, be2, (float*)d_out,
                                   (bf16_t*)nullptr, D);
}
